// LSTMClassifier_3745211483012
// MI455X (gfx1250) — matrix operand order repaired and checked by reading, not compiled, not run on hardware
//
#include <hip/hip_runtime.h>
#include <stdint.h>
#include <stddef.h>

// ---------------------------------------------------------------------------
// 2-layer LSTM (B=256, T=512, D=64, H=256) + final H->1 projection, CDNA5.
// Recurrent GEMMs run on v_wmma_f32_16x16x32_bf16 (bf16 in, f32 accumulate);
// cell state / activations / output stay f32. One tiny kernel per (layer,
// timestep) — stream order provides the device-wide dependency chain; hipGraph
// replay amortizes the 1024 launches. Step kernel is templated on the input
// width/stride so all fragment addresses fold to base + immediate offsets.
// ---------------------------------------------------------------------------

typedef __bf16 bf16_t;
typedef __attribute__((ext_vector_type(16))) __bf16 v16bf;
typedef __attribute__((ext_vector_type(8)))  __bf16 v8bf;
typedef __attribute__((ext_vector_type(8)))  float  v8f;

static constexpr int kB = 256;   // batch
static constexpr int kT = 512;   // seq len
static constexpr int kD = 64;    // input size
static constexpr int kH = 256;   // hidden
static constexpr int k4H = 4 * kH;

// ---------------- workspace layout (all offsets 256B aligned) ---------------
static constexpr size_t XB_OFF    = 0;                           // x as bf16: B*T*D*2 = 16,777,216
static constexpr size_t WCAT0_OFF = XB_OFF + 16777216;           // [4H, 64+256] bf16 = 655,360
static constexpr size_t WCAT1_OFF = WCAT0_OFF + 655360;          // [4H, 256+256] bf16 = 1,048,576
static constexpr size_t BIAS0_OFF = WCAT1_OFF + 1048576;         // 4H f32 = 4,096 (16,384 reserved)
static constexpr size_t BIAS1_OFF = BIAS0_OFF + 16384;           // 16,384
static constexpr size_t STATE_OFF = BIAS1_OFF + 16384;           // zeroed block below
static constexpr size_t C0_OFF    = STATE_OFF;                   // B*H f32 = 262,144
static constexpr size_t C1_OFF    = C0_OFF + 262144;             // 262,144
static constexpr size_t HB0_OFF   = C1_OFF + 262144;             // 2 x B*H bf16 = 262,144 (ping-pong)
static constexpr size_t HB1_OFF   = HB0_OFF + 262144;            // 262,144
static constexpr size_t H1F_OFF   = HB1_OFF + 262144;            // B*H f32 = 262,144
static constexpr size_t STATE_BYTES = (H1F_OFF + 262144) - STATE_OFF;  // 1,310,720
// total workspace use ~19.9 MB

// ------------------------------- setup kernels ------------------------------

__global__ void cvt_f32_to_bf16_kernel(const float* __restrict__ in,
                                       bf16_t* __restrict__ out, int n) {
  int i = blockIdx.x * blockDim.x + threadIdx.x;
  if (i < n) out[i] = (bf16_t)in[i];
}

// out[r, k] = (k < Din) ? Wih[r, k] : Whh[r, k - Din]   (bf16, ldw = Din + H)
__global__ void pack_wcat_kernel(const float* __restrict__ Wih,
                                 const float* __restrict__ Whh,
                                 int Din, bf16_t* __restrict__ out, int n) {
  int i = blockIdx.x * blockDim.x + threadIdx.x;
  if (i >= n) return;
  int ldw = Din + kH;
  int r = i / ldw;
  int k = i - r * ldw;
  float v = (k < Din) ? Wih[r * Din + k] : Whh[r * kH + (k - Din)];
  out[i] = (bf16_t)v;
}

__global__ void add_bias_kernel(const float* __restrict__ a,
                                const float* __restrict__ b,
                                float* __restrict__ o, int n) {
  int i = blockIdx.x * blockDim.x + threadIdx.x;
  if (i < n) o[i] = a[i] + b[i];
}

__global__ void zero_ws_kernel(uint32_t* __restrict__ p, int n_words) {
  int i = blockIdx.x * blockDim.x + threadIdx.x;
  if (i < n_words) p[i] = 0u;
}

// ------------------------------- LSTM step ----------------------------------
// grid = (16 batch-tiles, 16 hidden-tiles); block = 128 threads = 4 waves.
// Wave w computes gate w's 16x16 tile of z = [x_t | h_prev] @ Wcat^T + bias
// with chained v_wmma_f32_16x16x32_bf16, then LDS exchange + f32 cell update.
// DIN / STRIDE_A are compile-time so every fragment load is base + imm offset.

template <int DIN, int STRIDE_A, bool WRITE_F32>
__global__ __launch_bounds__(128)
void lstm_step_kernel(const bf16_t* __restrict__ Ain,   // input rows [B, DIN] (row stride STRIDE_A)
                      const bf16_t* __restrict__ hprev, // [B, H] bf16
                      const bf16_t* __restrict__ Wcat,  // [4H, DIN+H] bf16 row-major
                      const float*  __restrict__ bias,  // [4H]
                      float*  __restrict__ cstate,      // [B, H] f32 (in/out)
                      bf16_t* __restrict__ hb_out,      // [B, H] bf16
                      float*  __restrict__ hf_out) {    // [B, H] f32 (used if WRITE_F32)
  constexpr int LDW = DIN + kH;

  const int btile = blockIdx.x;        // 0..15
  const int htile = blockIdx.y;        // 0..15
  const int wave  = threadIdx.x >> 5;  // gate: 0=i 1=f 2=g 3=o
  const int lane  = threadIdx.x & 31;
  const int m     = lane & 15;         // A row within tile / B,C column
  const int g     = lane >> 4;         // lane group (ISA K-split)

  __shared__ float zTile[4][256];      // [gate][row*16 + col]

  // A-matrix source row (batch row) for this lane.
  const int brow = btile * 16 + m;
  const bf16_t* arow_x = Ain   + (size_t)brow * STRIDE_A;
  const bf16_t* arow_h = hprev + (size_t)brow * kH;

  // B-matrix column n == weight row wr (row-major weights == column-major B).
  const int wr = wave * kH + htile * 16 + m;
  const bf16_t* wrow = Wcat + (size_t)wr * LDW;

  v8f acc = {};

  // ---- phase 1: input projection, K = 0..DIN ----
#pragma unroll
  for (int k0 = 0; k0 < DIN; k0 += 32) {
    // A fragment (16x32 bf16): lane group g supplies K = 8g..8g+7 and
    // 16+8g..16+8g+7 within the chunk (ISA 7.12.2, 16-bit A layout).
    v8bf lo = *(const v8bf*)(arow_x + k0 + 8 * g);
    v8bf hi = *(const v8bf*)(arow_x + k0 + 16 + 8 * g);
    v16bf afrag;
#pragma unroll
    for (int i = 0; i < 8; ++i) { afrag[i] = lo[i]; afrag[8 + i] = hi[i]; }
    // B fragment (32x16 bf16): lane group g supplies K = 8g..8g+7 and
    // 16+8g..16+8g+7 within the chunk (same K order as A).
    v8bf blo = *(const v8bf*)(wrow + k0 + 8 * g);
    v8bf bhi = *(const v8bf*)(wrow + k0 + 16 + 8 * g);
    v16bf bfrag;
#pragma unroll
    for (int i = 0; i < 8; ++i) { bfrag[i] = blo[i]; bfrag[8 + i] = bhi[i]; }
    acc = __builtin_amdgcn_wmma_f32_16x16x32_bf16(
        false, afrag, false, bfrag, (short)0, acc, false, false);
  }

  // ---- phase 2: recurrent projection, K = DIN..DIN+H ----
#pragma unroll
  for (int k0 = 0; k0 < kH; k0 += 32) {
    v8bf lo = *(const v8bf*)(arow_h + k0 + 8 * g);
    v8bf hi = *(const v8bf*)(arow_h + k0 + 16 + 8 * g);
    v16bf afrag;
#pragma unroll
    for (int i = 0; i < 8; ++i) { afrag[i] = lo[i]; afrag[8 + i] = hi[i]; }
    v8bf blo = *(const v8bf*)(wrow + DIN + k0 + 8 * g);
    v8bf bhi = *(const v8bf*)(wrow + DIN + k0 + 16 + 8 * g);
    v16bf bfrag;
#pragma unroll
    for (int i = 0; i < 8; ++i) { bfrag[i] = blo[i]; bfrag[8 + i] = bhi[i]; }
    acc = __builtin_amdgcn_wmma_f32_16x16x32_bf16(
        false, afrag, false, bfrag, (short)0, acc, false, false);
  }

  // C/D layout: lane holds column n=m; VGPR j holds row j + 8g.  Bias is per
  // output column (weight row wr).
  const float bv = bias[wr];
#pragma unroll
  for (int j = 0; j < 8; ++j) {
    int row = j + 8 * g;
    zTile[wave][row * 16 + m] = acc[j] + bv;
  }
  __syncthreads();

  // Elementwise LSTM cell update (f32): 128 threads cover 256 tile elements.
#pragma unroll
  for (int e0 = 0; e0 < 256; e0 += 128) {
    const int e  = e0 + threadIdx.x;
    const int mm = e >> 4;
    const int nn = e & 15;
    const int b    = btile * 16 + mm;
    const int hcol = htile * 16 + nn;
    const size_t idx = (size_t)b * kH + hcol;
    float iv = zTile[0][e];
    float fv = zTile[1][e];
    float gv = zTile[2][e];
    float ov = zTile[3][e];
    iv = 1.0f / (1.0f + __expf(-iv));
    fv = 1.0f / (1.0f + __expf(-fv));
    gv = tanhf(gv);
    ov = 1.0f / (1.0f + __expf(-ov));
    const float cn = fv * cstate[idx] + iv * gv;
    const float hn = ov * tanhf(cn);
    cstate[idx] = cn;
    hb_out[idx] = (bf16_t)hn;
    if (WRITE_F32) hf_out[idx] = hn;
  }
}

// ----------------------------- final projection -----------------------------

__global__ void final_proj_kernel(const float* __restrict__ h1f,
                                  const float* __restrict__ Wf,
                                  const float* __restrict__ bf_,
                                  float* __restrict__ out) {
  int b = blockIdx.x * blockDim.x + threadIdx.x;
  if (b < kB) {
    float s = bf_[0];
#pragma unroll 4
    for (int k = 0; k < kH; ++k) s += h1f[(size_t)b * kH + k] * Wf[k];
    out[b] = s;
  }
}

// --------------------------------- launcher ---------------------------------

extern "C" void kernel_launch(void* const* d_in, const int* in_sizes, int n_in,
                              void* d_out, int out_size, void* d_ws, size_t ws_size,
                              hipStream_t stream) {
  (void)in_sizes; (void)n_in; (void)out_size; (void)ws_size;

  const float* x    = (const float*)d_in[0];
  const float* Wih0 = (const float*)d_in[1];
  const float* Whh0 = (const float*)d_in[2];
  const float* bih0 = (const float*)d_in[3];
  const float* bhh0 = (const float*)d_in[4];
  const float* Wih1 = (const float*)d_in[5];
  const float* Whh1 = (const float*)d_in[6];
  const float* bih1 = (const float*)d_in[7];
  const float* bhh1 = (const float*)d_in[8];
  const float* Wf   = (const float*)d_in[9];
  const float* bfp  = (const float*)d_in[10];

  char* ws = (char*)d_ws;
  bf16_t* xb    = (bf16_t*)(ws + XB_OFF);
  bf16_t* wcat0 = (bf16_t*)(ws + WCAT0_OFF);
  bf16_t* wcat1 = (bf16_t*)(ws + WCAT1_OFF);
  float*  bias0 = (float*)(ws + BIAS0_OFF);
  float*  bias1 = (float*)(ws + BIAS1_OFF);
  float*  c0    = (float*)(ws + C0_OFF);
  float*  c1    = (float*)(ws + C1_OFF);
  bf16_t* hb0   = (bf16_t*)(ws + HB0_OFF);   // 2 ping-pong buffers of B*H
  bf16_t* hb1   = (bf16_t*)(ws + HB1_OFF);
  float*  h1f   = (float*)(ws + H1F_OFF);

  // ---- setup (re-done every launch; deterministic) ----
  {
    const int nx = kB * kT * kD;                       // 8,388,608
    cvt_f32_to_bf16_kernel<<<nx / 256, 256, 0, stream>>>(x, xb, nx);

    const int n0 = k4H * (kD + kH);                    // 327,680
    pack_wcat_kernel<<<n0 / 256, 256, 0, stream>>>(Wih0, Whh0, kD, wcat0, n0);
    const int n1 = k4H * (kH + kH);                    // 524,288
    pack_wcat_kernel<<<n1 / 256, 256, 0, stream>>>(Wih1, Whh1, kH, wcat1, n1);

    add_bias_kernel<<<k4H / 256, 256, 0, stream>>>(bih0, bhh0, bias0, k4H);
    add_bias_kernel<<<k4H / 256, 256, 0, stream>>>(bih1, bhh1, bias1, k4H);

    const int nzw = (int)(STATE_BYTES / 4);            // 327,680 words
    zero_ws_kernel<<<nzw / 256, 256, 0, stream>>>((uint32_t*)(ws + STATE_OFF), nzw);
  }

  // ---- recurrence: 512 steps x 2 layers, stream-ordered ----
  const dim3 grid(16, 16, 1);
  const dim3 blk(128, 1, 1);
  const int hbStride = kB * kH;                        // elements per ping-pong buffer
  for (int t = 0; t < kT; ++t) {
    bf16_t* h0p = hb0 + (size_t)(t & 1) * hbStride;
    bf16_t* h0n = hb0 + (size_t)((t + 1) & 1) * hbStride;
    bf16_t* h1p = hb1 + (size_t)(t & 1) * hbStride;
    bf16_t* h1n = hb1 + (size_t)((t + 1) & 1) * hbStride;

    // layer 0: input row b is x[b, t, :]  (element stride T*D between rows)
    lstm_step_kernel<kD, kT * kD, false><<<grid, blk, 0, stream>>>(
        xb + (size_t)t * kD, h0p, wcat0, bias0, c0, h0n, h1f /*unused*/);
    // layer 1: input is layer 0's freshly written h (bf16)
    lstm_step_kernel<kH, kH, true><<<grid, blk, 0, stream>>>(
        h0n, h1p, wcat1, bias1, c1, h1n, h1f);
  }

  // ---- final projection: out[b] = h_last[b,:] . Wf + bf ----
  final_proj_kernel<<<1, 256, 0, stream>>>(h1f, Wf, bfp, (float*)d_out);
}
